// GE2ELoss_15659450761248
// MI455X (gfx1250) — compile-verified, matrix operand order checked by reading, not run on hardware
//
#include <hip/hip_runtime.h>

#define N_SPK 2048
#define G_UTT 10
#define D_DIM 256
#define NG (N_SPK * G_UTT)       // 20480 rows
#define NTILE (N_SPK / 16)       // 128 column tiles
#define MAIN_BLOCKS 160          // 160 blocks * 8 waves * 16 rows = 20480
#define EPS_COS 1e-8f
#define EPS_CLAMP 1e-6f
#define FIXMAX 8.0f              // logit <= w*1+b = 5 < 8 (safe fixed softmax max)
#define LDS_ROW 264              // 256 + 8 ushort pad -> conflict-free b128 reads

typedef __attribute__((ext_vector_type(16))) __bf16 v16bf;
typedef __attribute__((ext_vector_type(8)))  float  v8f;

struct U4 { unsigned int x, y, z, w; };
union V16 { v16bf v; U4 q[2]; };

__device__ __forceinline__ unsigned short f2bf(float f) {
  unsigned u = __float_as_uint(f);
  u += 0x7FFFu + ((u >> 16) & 1u);      // round-to-nearest-even
  return (unsigned short)(u >> 16);
}

// ---------------------------------------------------------------------------
// Kernel A: per-speaker stats, exact fp32 diagonal, bf16 packing.
// One block per speaker n; thread d owns feature element d (D=256).
// ---------------------------------------------------------------------------
__global__ __launch_bounds__(256) void ge2e_stats(
    const float* __restrict__ x,
    unsigned short* __restrict__ xbf,
    unsigned short* __restrict__ centbf,
    float* __restrict__ invxn,
    float* __restrict__ invcn,
    float* __restrict__ cosd)
{
  const int n = blockIdx.x;
  const int d = threadIdx.x;
  const float* xr = x + (size_t)n * G_UTT * D_DIM + d;

  float xg[G_UTT];
  float s = 0.f;
  #pragma unroll
  for (int g = 0; g < G_UTT; ++g) { xg[g] = xr[g * D_DIM]; s += xg[g]; }

  centbf[n * D_DIM + d] = f2bf(s * (1.0f / G_UTT));
  #pragma unroll
  for (int g = 0; g < G_UTT; ++g)
    xbf[((size_t)n * G_UTT + g) * D_DIM + d] = f2bf(xg[g]);

  float xx[G_UTT], xs[G_UTT];
  #pragma unroll
  for (int g = 0; g < G_UTT; ++g) { xx[g] = xg[g] * xg[g]; xs[g] = xg[g] * s; }
  float ss = s * s;

  #pragma unroll
  for (int off = 16; off; off >>= 1) {
    #pragma unroll
    for (int g = 0; g < G_UTT; ++g) {
      xx[g] += __shfl_xor(xx[g], off, 32);
      xs[g] += __shfl_xor(xs[g], off, 32);
    }
    ss += __shfl_xor(ss, off, 32);
  }

  __shared__ float sh[21];
  if (threadIdx.x < 21) sh[threadIdx.x] = 0.f;
  __syncthreads();
  if ((threadIdx.x & 31) == 0) {
    #pragma unroll
    for (int g = 0; g < G_UTT; ++g) {
      atomicAdd(&sh[g], xx[g]);
      atomicAdd(&sh[10 + g], xs[g]);
    }
    atomicAdd(&sh[20], ss);
  }
  __syncthreads();

  if (threadIdx.x < G_UTT) {
    const int g = threadIdx.x;
    const float XX = sh[g], XS = sh[10 + g], SS = sh[20];
    const float xn = fmaxf(sqrtf(XX), EPS_COS);
    const float e2 = fmaxf(SS - 2.f * XS + XX, 0.f);
    const float en = fmaxf(sqrtf(e2) * (1.f / (G_UTT - 1)), EPS_COS);
    cosd[n * G_UTT + g]  = ((XS - XX) * (1.f / (G_UTT - 1))) / (xn * en);
    invxn[n * G_UTT + g] = 1.f / xn;
  }
  if (threadIdx.x == 0) {
    const float cn = fmaxf(sqrtf(sh[20]) * (1.0f / G_UTT), EPS_COS);
    invcn[n] = 1.f / cn;
  }
}

// ---------------------------------------------------------------------------
// Kernel B: fused bf16-WMMA GEMM + fixed-max softmax/argmax.
// B tiles staged block-wide into LDS via async-to-LDS (ASYNCcnt), double
// buffered; A tile (16x256 bf16) lives in 64 VGPRs per wave.
// ---------------------------------------------------------------------------
__global__ __launch_bounds__(256) void ge2e_main(
    const unsigned short* __restrict__ xbf,
    const unsigned short* __restrict__ centbf,
    const float* __restrict__ invxn,
    const float* __restrict__ invcn,
    const float* __restrict__ cosd,
    const float* __restrict__ wp,
    const float* __restrict__ bp,
    float* __restrict__ ploss,
    float* __restrict__ pcorr)
{
  __shared__ unsigned short sB[2][16 * LDS_ROW];   // 2 x 8.25 KB padded tiles
  __shared__ float s_loss, s_corr;
  if (threadIdx.x == 0) { s_loss = 0.f; s_corr = 0.f; }

  const int tid  = threadIdx.x;
  const int lane = tid & 31;
  const int wave = tid >> 5;
  const int row_block = (blockIdx.x * 8 + wave) * 16;
  const int lm  = lane & 15;
  const int sel = lane >> 4;
  const float W = *wp, Bb = *bp;

  // staging map: thread copies 32B; src row-major 16x256, dst padded 16x264
  const int st_row = tid >> 4;                   // 0..15 tile row (= centroid)
  const int st_c16 = tid & 15;                   // 16-ushort chunk in row
  const int st_src = st_row * 256 + st_c16 * 16; // ushort offset in tile
  const int st_dst = st_row * LDS_ROW + st_c16 * 16;

  // issue async copy of tile t into buffer (t&1): 2 x b128 per thread
  #define ISSUE_TILE(t)                                                       \
    do {                                                                      \
      const unsigned short* g_ = centbf + (size_t)(t) * 4096 + st_src;        \
      unsigned l_ = (unsigned)(unsigned long long)&sB[(t) & 1][st_dst];       \
      asm volatile("global_load_async_to_lds_b128 %0, %1, off"                \
                   :: "v"(l_), "v"(g_) : "memory");                           \
      asm volatile("global_load_async_to_lds_b128 %0, %1, off offset:16"      \
                   :: "v"(l_), "v"(g_) : "memory");                           \
    } while (0)

  // preload A tile: 16 rows x 256 K of bf16 (ISA 16-bit A layout)
  V16 A[8];
  {
    const unsigned short* xr = xbf + (size_t)(row_block + lm) * D_DIM;
    #pragma unroll
    for (int c = 0; c < 8; ++c) {
      const int kb = c * 32;
      A[c].q[0] = *(const U4*)(xr + kb + sel * 8);
      A[c].q[1] = *(const U4*)(xr + kb + 16 + sel * 8);
    }
  }

  // this lane's 8 rows (C/D layout): m = v + 8*sel
  float invx[8], cdv[8];
  int   labcol[8];
  #pragma unroll
  for (int v = 0; v < 8; ++v) {
    const int R = row_block + v + 8 * sel;
    invx[v]   = invxn[R];
    cdv[v]    = cosd[R];
    labcol[v] = R / G_UTT;
  }

  float sm[8], bestv[8], labv[8];
  int   besti[8];
  #pragma unroll
  for (int v = 0; v < 8; ++v) {
    sm[v] = 0.f; bestv[v] = -3.4e38f; labv[v] = -3.4e38f; besti[v] = 0;
  }

  ISSUE_TILE(0);

  for (int t = 0; t < NTILE; ++t) {
    asm volatile("s_wait_asynccnt 0x0" ::: "memory");  // my tile-t portion done
    __syncthreads();                                   // everyone's portion done
    if (t + 1 < NTILE) ISSUE_TILE(t + 1);              // overlap next copy

    const unsigned short* bs = &sB[t & 1][lm * LDS_ROW];
    v8f acc = {};
    #pragma unroll
    for (int c = 0; c < 8; ++c) {
      const int kb = c * 32;
      V16 Bm;
      Bm.q[0] = *(const U4*)(bs + kb + sel * 8);        // ds_load_b128
      Bm.q[1] = *(const U4*)(bs + kb + 16 + sel * 8);
      acc = __builtin_amdgcn_wmma_f32_16x16x32_bf16(
          false, A[c].v, false, Bm.v, (short)0, acc, false, false);
    }

    const int j = t * 16 + lm;
    const float icn = invcn[j];
    #pragma unroll
    for (int v = 0; v < 8; ++v) {
      float val = acc[v] * (invx[v] * icn);
      if (j == labcol[v]) val = cdv[v];                 // exclusive-centroid diag
      val = fmaxf(val, EPS_CLAMP);
      const float logit = fmaf(val, W, Bb);
      if (j == labcol[v]) labv[v] = logit;
      if (logit > bestv[v]) { bestv[v] = logit; besti[v] = j; }
      sm[v] += __expf(logit - FIXMAX);                  // fixed-max softmax
    }
  }

  // merge across the 16 lanes of each half-wave (rows are per-half)
  float lsum = 0.f, csum = 0.f;
  #pragma unroll
  for (int v = 0; v < 8; ++v) {
    #pragma unroll
    for (int off = 1; off < 16; off <<= 1) {
      sm[v] += __shfl_xor(sm[v], off, 32);
      const float bo = __shfl_xor(bestv[v], off, 32);
      const int   io = __shfl_xor(besti[v], off, 32);
      if (bo > bestv[v] || (bo == bestv[v] && io < besti[v])) {
        bestv[v] = bo; besti[v] = io;
      }
      labv[v] = fmaxf(labv[v], __shfl_xor(labv[v], off, 32));
    }
    if (lm == 0) {
      lsum += (FIXMAX + __logf(sm[v])) - labv[v];       // logsumexp - label
      csum += (besti[v] == labcol[v]) ? 1.f : 0.f;
    }
  }
  if (lm == 0) { atomicAdd(&s_loss, lsum); atomicAdd(&s_corr, csum); }
  __syncthreads();
  if (threadIdx.x == 0) {
    ploss[blockIdx.x] = s_loss;
    pcorr[blockIdx.x] = s_corr;
  }
  #undef ISSUE_TILE
}

// ---------------------------------------------------------------------------
// Kernel C: final reduction over 160 block partials -> [nloss, prec1]
// ---------------------------------------------------------------------------
__global__ __launch_bounds__(256) void ge2e_final(
    const float* __restrict__ ploss,
    const float* __restrict__ pcorr,
    int nblocks,
    float* __restrict__ out)
{
  const int t = threadIdx.x;
  float l = (t < nblocks) ? ploss[t] : 0.f;
  float c = (t < nblocks) ? pcorr[t] : 0.f;
  #pragma unroll
  for (int off = 16; off; off >>= 1) {
    l += __shfl_xor(l, off, 32);
    c += __shfl_xor(c, off, 32);
  }
  __shared__ float sl, sc;
  if (t == 0) { sl = 0.f; sc = 0.f; }
  __syncthreads();
  if ((t & 31) == 0) { atomicAdd(&sl, l); atomicAdd(&sc, c); }
  __syncthreads();
  if (t == 0) {
    out[0] = sl / (float)NG;
    out[1] = 100.f * sc / (float)NG;
  }
}

// ---------------------------------------------------------------------------
// Workspace layout (bytes):
//   xbf   : 20480*256*2 = 10485760   @ 0
//   centbf:  2048*256*2 =  1048576   @ 10485760
//   invxn : 20480*4     =    81920   @ 11534336
//   invcn :  2048*4     =     8192   @ 11616256
//   cosd  : 20480*4     =    81920   @ 11624448
//   ploss :   160*4     =      640   @ 11706368
//   pcorr :   160*4     =      640   @ 11707008
// ---------------------------------------------------------------------------
extern "C" void kernel_launch(void* const* d_in, const int* in_sizes, int n_in,
                              void* d_out, int out_size, void* d_ws, size_t ws_size,
                              hipStream_t stream)
{
  const float* x = (const float*)d_in[0];
  const float* w = (const float*)d_in[1];
  const float* b = (const float*)d_in[2];
  (void)in_sizes; (void)n_in; (void)out_size; (void)ws_size;

  char* ws = (char*)d_ws;
  unsigned short* xbf   = (unsigned short*)(ws + 0);
  unsigned short* cbf   = (unsigned short*)(ws + 10485760);
  float*          invxn = (float*)(ws + 11534336);
  float*          invcn = (float*)(ws + 11616256);
  float*          cosd  = (float*)(ws + 11624448);
  float*          ploss = (float*)(ws + 11706368);
  float*          pcorr = (float*)(ws + 11707008);

  hipLaunchKernelGGL(ge2e_stats, dim3(N_SPK), dim3(256), 0, stream,
                     x, xbf, cbf, invxn, invcn, cosd);
  hipLaunchKernelGGL(ge2e_main, dim3(MAIN_BLOCKS), dim3(256), 0, stream,
                     xbf, cbf, invxn, invcn, cosd, w, b, ploss, pcorr);
  hipLaunchKernelGGL(ge2e_final, dim3(1), dim3(256), 0, stream,
                     ploss, pcorr, MAIN_BLOCKS, (float*)d_out);
}
